// CrossAttention_89670327206357
// MI455X (gfx1250) — matrix operand order repaired and checked by reading, not compiled, not run on hardware
//
#include <hip/hip_runtime.h>
#include <hip/hip_bf16.h>

// CrossAttention for MI455X (gfx1250, wave32, WMMA f32_16x16x32_f16).
// r3: GEMM staging software-pipelined: next-tile global loads issued BEFORE
//     the WMMA block, converts/LDS-stores after it, so load latency hides
//     behind matrix math (prev revision serialized load->wait->cvt->store).

typedef __attribute__((ext_vector_type(16))) _Float16 v16h;
typedef __attribute__((ext_vector_type(8)))  _Float16 v8h;
typedef __attribute__((ext_vector_type(8)))  float    v8f;

#define B_   4
#define T_   2048
#define HD_  1024      // hidden dim (== Q_DIM == C_DIM == OUT_DIM)
#define NH_  16
#define DH_  64
#define MTOT (B_*T_)   // 8192 tokens

static __device__ inline v16h cat8(v8h lo, v8h hi) {
  return __builtin_shufflevector(lo, hi, 0,1,2,3,4,5,6,7,8,9,10,11,12,13,14,15);
}

static __device__ inline v8f wmma16(v16h a, v16h b, v8f c) {
  // D = A(16x32 f16) * B(32x16 f16) + C(16x16 f32)
  return __builtin_amdgcn_wmma_f32_16x16x32_f16(false, a, false, b, (short)0, c,
                                                false, false);
}

// Wave-local LDS ordering fence: LDS ops from one wave are processed in order
// (ISA 7.3); this stops compiler reordering and drains DScnt. No block barrier.
static __device__ inline void lds_fence_wave() {
  asm volatile("s_wait_dscnt 0" ::: "memory");
}

// ---------------------------------------------------------------------------
// GEMM: Y[f16] = X[f32, MTOTxHD_] @ W[f32, HD_xHD_] + bias.
// vmode 0: Y layout [b][h][t][d] (Q, K).  vmode 1: Y layout [b][h][d][t] (V^T).
// Block 256 thr = 8 waves; tile 128(M) x 128(N); K-step 32 double-buffered;
// wave tile 32x64 -> 8 WMMAs per K-step; staging pipelined around the WMMAs.
// ---------------------------------------------------------------------------
__global__ __launch_bounds__(256) void gemm_proj_kernel(
    const float* __restrict__ X, const float* __restrict__ W,
    const float* __restrict__ bias, _Float16* __restrict__ Y, int vmode)
{
  __shared__ _Float16 Xs[2][128][40];   // [buf][row][k], pitch 80B
  __shared__ _Float16 Ws[2][128][40];   // [buf][n][k] (transposed)

  const int tid  = threadIdx.x;
  const int lane = tid & 31;
  const int wv   = tid >> 5;
  const int wm   = wv & 3;              // wave M block (32 rows)
  const int wn   = wv >> 2;             // wave N block (64 cols)
  const int m0   = blockIdx.x * 128;
  const int n0   = blockIdx.y * 128;
  const int l16  = lane & 15;
  const int hg   = lane >> 4;

  const int xr  = tid >> 3;             // X stage: row 0..31 (+32*i)
  const int xc  = (tid & 7) << 2;       // X stage: k col 0..28
  const int wk  = tid >> 3;             // W stage: k row 0..31
  const int wnb = (tid & 7) << 4;       // W stage: n base 0..112 (+4*i)

  auto ld_tile = [&](float4* tx, float4* tw, int k0) {
    #pragma unroll
    for (int i = 0; i < 4; ++i)
      tx[i] = *reinterpret_cast<const float4*>(
          X + (size_t)(m0 + xr + 32 * i) * HD_ + k0 + xc);
    #pragma unroll
    for (int i = 0; i < 4; ++i)
      tw[i] = *reinterpret_cast<const float4*>(
          W + (size_t)(k0 + wk) * HD_ + n0 + wnb + 4 * i);
  };
  auto st_tile = [&](int buf, const float4* tx, const float4* tw) {
    #pragma unroll
    for (int i = 0; i < 4; ++i) {
      const int row = xr + 32 * i;
      Xs[buf][row][xc + 0] = (_Float16)tx[i].x;
      Xs[buf][row][xc + 1] = (_Float16)tx[i].y;
      Xs[buf][row][xc + 2] = (_Float16)tx[i].z;
      Xs[buf][row][xc + 3] = (_Float16)tx[i].w;
    }
    #pragma unroll
    for (int i = 0; i < 4; ++i) {
      const int nf = wnb + 4 * i;
      Ws[buf][nf + 0][wk] = (_Float16)tw[i].x;
      Ws[buf][nf + 1][wk] = (_Float16)tw[i].y;
      Ws[buf][nf + 2][wk] = (_Float16)tw[i].z;
      Ws[buf][nf + 3][wk] = (_Float16)tw[i].w;
    }
  };

  v8f zero = {};
  v8f acc[2][4];
  #pragma unroll
  for (int i = 0; i < 2; ++i)
    #pragma unroll
    for (int j = 0; j < 4; ++j) acc[i][j] = zero;

  {
    float4 tx[4], tw[4];
    ld_tile(tx, tw, 0);
    st_tile(0, tx, tw);
  }
  __syncthreads();

  for (int k0 = 0; k0 < HD_; k0 += 32) {
    const int cur  = (k0 >> 5) & 1;
    const bool more = (k0 + 32 < HD_);

    // (1) issue next-tile global loads (no waits yet)
    float4 tx[4], tw[4];
    if (more) ld_tile(tx, tw, k0 + 32);

    // (2) compute current tile: fragment loads + 8 WMMAs hide load latency
    v16h af[2], bf[4];
    #pragma unroll
    for (int i = 0; i < 2; ++i) {
      const int row = wm * 32 + i * 16 + l16;
      af[i] = cat8(*(const v8h*)&Xs[cur][row][hg * 16],
                   *(const v8h*)&Xs[cur][row][hg * 16 + 8]);
    }
    #pragma unroll
    for (int j = 0; j < 4; ++j) {
      const int col = wn * 64 + j * 16 + l16;
      bf[j] = cat8(*(const v8h*)&Ws[cur][col][hg * 16],
                   *(const v8h*)&Ws[cur][col][hg * 16 + 8]);
    }
    #pragma unroll
    for (int i = 0; i < 2; ++i)
      #pragma unroll
      for (int j = 0; j < 4; ++j)
        acc[i][j] = wmma16(af[i], bf[j], acc[i][j]);

    // (3) convert + store next tile into the other buffer
    if (more) st_tile(cur ^ 1, tx, tw);
    __syncthreads();
  }

  // --- epilogue: bias, f16 convert, head-split scatter ---
  #pragma unroll
  for (int i = 0; i < 2; ++i) {
    #pragma unroll
    for (int j = 0; j < 4; ++j) {
      const int ncol = n0 + wn * 64 + j * 16 + l16;
      const float bv = bias[ncol];
      const int h = ncol >> 6, d = ncol & 63;
      #pragma unroll
      for (int r = 0; r < 8; ++r) {
        const int m = m0 + wm * 32 + i * 16 + hg * 8 + r;   // C layout row
        const int b = m >> 11, t = m & (T_ - 1);
        const float val = acc[i][j][r] + bv;
        size_t addr;
        if (vmode == 0) addr = ((size_t)(b * NH_ + h) * T_ + t) * DH_ + d;
        else            addr = ((size_t)(b * NH_ + h) * DH_ + d) * T_ + t;
        Y[addr] = (_Float16)val;
      }
    }
  }
}

// ---------------------------------------------------------------------------
// Flash attention: one wave owns 32 query rows (2 row-subtiles); kv swept in
// steps of 32. K/V fragments shared across row-subtiles. Waves independent.
// Q [bh][t][d], K [bh][t][d], V^T [bh][d][t] f16 -> attn [b][t][h*64+d] f16.
// ---------------------------------------------------------------------------
__global__ __launch_bounds__(128) void flash_attn_kernel(
    const _Float16* __restrict__ Qg, const _Float16* __restrict__ Kg,
    const _Float16* __restrict__ Vt, _Float16* __restrict__ attn)
{
  __shared__ _Float16 Ps[4][16][40];   // per-wave P subtile [q row][kv]

  const int lane = threadIdx.x & 31;
  const int wv   = threadIdx.x >> 5;
  const int tile = blockIdx.x * 4 + wv;        // 0 .. B*NH*(T/32)-1
  const int qt   = tile & (T_ / 32 - 1);       // 32-row q tile within head
  const int bh   = tile >> 6;                  // b*NH + h
  const int b    = bh >> 4, h = bh & 15;
  const int l16  = lane & 15;
  const int hg   = lane >> 4;

  const _Float16* Qp = Qg + ((size_t)bh * T_ + qt * 32) * DH_;
  const _Float16* Kp = Kg + (size_t)bh * T_ * DH_;
  const _Float16* Vp = Vt + (size_t)bh * DH_ * T_;

  // Q in registers for the whole sweep: 2 row-subtiles x 2 d-steps.
  v16h qf[2][2];
  #pragma unroll
  for (int i = 0; i < 2; ++i)
    #pragma unroll
    for (int j = 0; j < 2; ++j) {
      const _Float16* qrow = Qp + (size_t)(i * 16 + l16) * DH_ + j * 32 + hg * 16;
      qf[i][j] = cat8(*(const v8h*)qrow, *(const v8h*)(qrow + 8));
    }

  v8f zero = {};
  v8f oacc[2][4];
  #pragma unroll
  for (int i = 0; i < 2; ++i)
    #pragma unroll
    for (int d = 0; d < 4; ++d) oacc[i][d] = zero;
  float mrow[2][8], lrow[2][8];
  #pragma unroll
  for (int i = 0; i < 2; ++i)
    #pragma unroll
    for (int r = 0; r < 8; ++r) { mrow[i][r] = -1e30f; lrow[i][r] = 0.0f; }
  const float scale = 0.125f;        // 1/sqrt(64)

  for (int kv = 0; kv < T_; kv += 32) {
    // K fragments (shared by both row-subtiles): lane col = kv token,
    // 16 contiguous d.  V^T fragments: lane col = d, 16 contiguous kv.
    v16h kf[2][2], vf[4];
    #pragma unroll
    for (int ct = 0; ct < 2; ++ct)
      #pragma unroll
      for (int j = 0; j < 2; ++j)
        kf[ct][j] = *(const v16h*)(
            Kp + (size_t)(kv + ct * 16 + l16) * DH_ + j * 32 + hg * 16);
    #pragma unroll
    for (int ds = 0; ds < 4; ++ds)
      vf[ds] = *(const v16h*)(
          Vp + (size_t)(ds * 16 + l16) * T_ + kv + hg * 16);

    if (kv + 32 < T_) {   // prime L2/L0 for the next kv step
      __builtin_prefetch((const void*)(
          Kp + (size_t)(kv + 32 + l16) * DH_ + hg * 32), 0, 1);
      __builtin_prefetch((const void*)(
          Vp + (size_t)(hg * 16 + l16) * T_ + kv + 32), 0, 1);
    }

    #pragma unroll
    for (int i = 0; i < 2; ++i) {
      // ---- S = Q K^T for this 16q x 32kv slab
      v8f st[2];
      #pragma unroll
      for (int ct = 0; ct < 2; ++ct) {
        v8f s = wmma16(qf[i][0], kf[ct][0], zero);
        st[ct] = wmma16(qf[i][1], kf[ct][1], s);
      }

      // ---- online softmax (each row lives in one 16-lane group)
      float mnew[8], alpha[8];
      #pragma unroll
      for (int r = 0; r < 8; ++r) {
        float a = st[0][r] * scale, c = st[1][r] * scale;
        st[0][r] = a; st[1][r] = c;
        float mx = fmaxf(a, c);
        #pragma unroll
        for (int msk = 1; msk <= 8; msk <<= 1)
          mx = fmaxf(mx, __shfl_xor(mx, msk, 32));
        mnew[r]     = fmaxf(mrow[i][r], mx);
        alpha[r]    = __expf(mrow[i][r] - mnew[r]);
        mrow[i][r]  = mnew[r];
      }
      #pragma unroll
      for (int r = 0; r < 8; ++r) {
        float p0 = __expf(st[0][r] - mnew[r]);
        float p1 = __expf(st[1][r] - mnew[r]);
        st[0][r] = p0; st[1][r] = p1;
        float ps = p0 + p1;
        #pragma unroll
        for (int msk = 1; msk <= 8; msk <<= 1) ps += __shfl_xor(ps, msk, 32);
        lrow[i][r] = lrow[i][r] * alpha[r] + ps;
      }
      #pragma unroll
      for (int ds = 0; ds < 4; ++ds)
        #pragma unroll
        for (int r = 0; r < 8; ++r) oacc[i][ds][r] *= alpha[r];

      // ---- reshape P: C layout -> per-wave LDS [q][kv] -> A fragment.
      lds_fence_wave();                       // prior reads issued before writes
      #pragma unroll
      for (int ct = 0; ct < 2; ++ct)
        #pragma unroll
        for (int r = 0; r < 8; ++r)
          Ps[wv][hg * 8 + r][ct * 16 + l16] = (_Float16)st[ct][r];
      lds_fence_wave();                       // writes issued before reads
      const _Float16* prow = &Ps[wv][l16][hg * 16];
      v16h pf = cat8(*(const v8h*)prow, *(const v8h*)(prow + 8));

      // ---- O += P * V
      #pragma unroll
      for (int ds = 0; ds < 4; ++ds)
        oacc[i][ds] = wmma16(pf, vf[ds], oacc[i][ds]);
    }
  }

  // ---- normalize and store attn[b][t][h*64+d] as f16
  #pragma unroll
  for (int i = 0; i < 2; ++i)
    #pragma unroll
    for (int ds = 0; ds < 4; ++ds) {
      const int d = ds * 16 + l16;
      #pragma unroll
      for (int r = 0; r < 8; ++r) {
        const int t = qt * 32 + i * 16 + hg * 8 + r;
        const float o = oacc[i][ds][r] / lrow[i][r];
        attn[((size_t)b * T_ + t) * HD_ + h * 64 + d] = (_Float16)o;
      }
    }
}

// ---------------------------------------------------------------------------
// Output GEMM: out[f32] = A[f16, MTOTxHD_] @ Wo[f32] + bo. Same tiling scheme.
// ---------------------------------------------------------------------------
__global__ __launch_bounds__(256) void gemm_out_kernel(
    const _Float16* __restrict__ X, const float* __restrict__ W,
    const float* __restrict__ bias, float* __restrict__ out)
{
  __shared__ _Float16 Xs[2][128][40];
  __shared__ _Float16 Ws[2][128][40];

  const int tid  = threadIdx.x;
  const int lane = tid & 31;
  const int wv   = tid >> 5;
  const int wm   = wv & 3;
  const int wn   = wv >> 2;
  const int m0   = blockIdx.x * 128;
  const int n0   = blockIdx.y * 128;
  const int l16  = lane & 15;
  const int hg   = lane >> 4;

  const int xr  = tid >> 2;             // 0..63 (+64*i)
  const int xc  = (tid & 3) << 3;       // 0,8,16,24
  const int wk  = tid >> 3;             // 0..31
  const int wnb = (tid & 7) << 4;       // 0..112

  auto ld_tile = [&](v8h* tx, float4* tw, int k0) {
    #pragma unroll
    for (int i = 0; i < 2; ++i)
      tx[i] = *(const v8h*)(X + (size_t)(m0 + xr + 64 * i) * HD_ + k0 + xc);
    #pragma unroll
    for (int i = 0; i < 4; ++i)
      tw[i] = *reinterpret_cast<const float4*>(
          W + (size_t)(k0 + wk) * HD_ + n0 + wnb + 4 * i);
  };
  auto st_tile = [&](int buf, const v8h* tx, const float4* tw) {
    #pragma unroll
    for (int i = 0; i < 2; ++i)
      *(v8h*)&Xs[buf][xr + 64 * i][xc] = tx[i];
    #pragma unroll
    for (int i = 0; i < 4; ++i) {
      const int nf = wnb + 4 * i;
      Ws[buf][nf + 0][wk] = (_Float16)tw[i].x;
      Ws[buf][nf + 1][wk] = (_Float16)tw[i].y;
      Ws[buf][nf + 2][wk] = (_Float16)tw[i].z;
      Ws[buf][nf + 3][wk] = (_Float16)tw[i].w;
    }
  };

  v8f zero = {};
  v8f acc[2][4];
  #pragma unroll
  for (int i = 0; i < 2; ++i)
    #pragma unroll
    for (int j = 0; j < 4; ++j) acc[i][j] = zero;

  {
    v8h tx[2]; float4 tw[4];
    ld_tile(tx, tw, 0);
    st_tile(0, tx, tw);
  }
  __syncthreads();

  for (int k0 = 0; k0 < HD_; k0 += 32) {
    const int cur  = (k0 >> 5) & 1;
    const bool more = (k0 + 32 < HD_);

    v8h tx[2]; float4 tw[4];
    if (more) ld_tile(tx, tw, k0 + 32);

    v16h af[2], bf[4];
    #pragma unroll
    for (int i = 0; i < 2; ++i) {
      const int row = wm * 32 + i * 16 + l16;
      af[i] = cat8(*(const v8h*)&Xs[cur][row][hg * 16],
                   *(const v8h*)&Xs[cur][row][hg * 16 + 8]);
    }
    #pragma unroll
    for (int j = 0; j < 4; ++j) {
      const int col = wn * 64 + j * 16 + l16;
      bf[j] = cat8(*(const v8h*)&Ws[cur][col][hg * 16],
                   *(const v8h*)&Ws[cur][col][hg * 16 + 8]);
    }
    #pragma unroll
    for (int i = 0; i < 2; ++i)
      #pragma unroll
      for (int j = 0; j < 4; ++j)
        acc[i][j] = wmma16(af[i], bf[j], acc[i][j]);

    if (more) st_tile(cur ^ 1, tx, tw);
    __syncthreads();
  }

  #pragma unroll
  for (int i = 0; i < 2; ++i) {
    #pragma unroll
    for (int j = 0; j < 4; ++j) {
      const int ncol = n0 + wn * 64 + j * 16 + l16;
      const float bv = bias[ncol];
      #pragma unroll
      for (int r = 0; r < 8; ++r) {
        const int m = m0 + wm * 32 + i * 16 + hg * 8 + r;
        out[(size_t)m * HD_ + ncol] = acc[i][j][r] + bv;
      }
    }
  }
}

// ---------------------------------------------------------------------------
extern "C" void kernel_launch(void* const* d_in, const int* in_sizes, int n_in,
                              void* d_out, int out_size, void* d_ws, size_t ws_size,
                              hipStream_t stream) {
  const float* query   = (const float*)d_in[0];
  const float* context = (const float*)d_in[1];
  const float* Wq = (const float*)d_in[2];
  const float* bq = (const float*)d_in[3];
  const float* Wk = (const float*)d_in[4];
  const float* bk = (const float*)d_in[5];
  const float* Wv = (const float*)d_in[6];
  const float* bv = (const float*)d_in[7];
  const float* Wo = (const float*)d_in[8];
  const float* bo = (const float*)d_in[9];

  // Workspace: Q, K (head-major), V^T, attn — 4 x 16 MB of f16.
  _Float16* Qh  = (_Float16*)d_ws;
  _Float16* Kh  = Qh  + (size_t)MTOT * HD_;
  _Float16* Vth = Kh  + (size_t)MTOT * HD_;
  _Float16* Ah  = Vth + (size_t)MTOT * HD_;

  dim3 g(MTOT / 128, HD_ / 128);   // 64 x 8
  gemm_proj_kernel<<<g, 256, 0, stream>>>(query,   Wq, bq, Qh,  0);
  gemm_proj_kernel<<<g, 256, 0, stream>>>(context, Wk, bk, Kh,  0);
  gemm_proj_kernel<<<g, 256, 0, stream>>>(context, Wv, bv, Vth, 1);

  const int n_qtiles = B_ * NH_ * (T_ / 32);       // 4096 wave tiles
  flash_attn_kernel<<<n_qtiles / 4, 128, 0, stream>>>(Qh, Kh, Vth, Ah);

  gemm_out_kernel<<<g, 256, 0, stream>>>(Ah, Wo, bo, (float*)d_out);
}
